// MegNetLayer_81570018885993
// MI455X (gfx1250) — compile-verified, matrix operand order checked by reading, not run on hardware
//
#include <hip/hip_runtime.h>
#include <hip/hip_bf16.h>

typedef __attribute__((ext_vector_type(16))) _Float16 v16h;
typedef __attribute__((ext_vector_type(8)))  _Float16 h8;
typedef __attribute__((ext_vector_type(8)))  float    v8f;

#define BATCH 4
#define NB 262144
#define NA 8192
#define FDIM 32
#define TB 256

__device__ __forceinline__ float softplus_f(float x) {
    // numerically stable log1p(exp(x))
    return fmaxf(x, 0.0f) + log1pf(expf(-fabsf(x)));
}

// Build a 16-half WMMA fragment from two contiguous 8-half runs (ISA 7.12.2:
// per-lane K order is k0..k0+7 then k0+16..k0+23).
__device__ __forceinline__ v16h load_frag(const _Float16* p) {
    h8 lo = *(const h8*)(p);
    h8 hi = *(const h8*)(p + 16);
    return __builtin_shufflevector(lo, hi, 0,1,2,3,4,5,6,7,8,9,10,11,12,13,14,15);
}

__global__ void zero_kernel(float* p, unsigned long long n) {
    unsigned long long i = (unsigned long long)blockIdx.x * TB + threadIdx.x;
    if (i < n) p[i] = 0.0f;
}

// ---------------------------------------------------------------------------
// Kernel 1: fused gather -> edge MLP (128->64->64->32) -> scatter-add to atoms
// 128 bonds per block (8 waves x 16-row M-tiles), WMMA f32_16x16x32_f16.
// ---------------------------------------------------------------------------
__global__ void __launch_bounds__(TB)
bond_kernel(const float* __restrict__ bonds, const int* __restrict__ ba1,
            const int* __restrict__ ba2, const float* __restrict__ atoms,
            const float* __restrict__ state,
            const float* __restrict__ w1, const float* __restrict__ b1,
            const float* __restrict__ w2, const float* __restrict__ b2,
            const float* __restrict__ w3, const float* __restrict__ b3,
            float* __restrict__ out_bonds, float* __restrict__ accA,
            float* __restrict__ cnt, float* __restrict__ bond_sum)
{
    __shared__ int s_i1[128];
    __shared__ int s_i2[128];
    __shared__ __align__(16) _Float16 s_ein[128 * 128];  // e_in tile (f16)
    __shared__ __align__(16) _Float16 s_wt1[64 * 128];   // w1^T [n][k]
    __shared__ __align__(16) _Float16 s_wt2[64 * 64];    // w2^T
    __shared__ __align__(16) _Float16 s_wt3[32 * 64];    // w3^T
    __shared__ __align__(16) _Float16 s_act[128 * 64];   // interlayer acts
    __shared__ float s_b1[64], s_b2[64], s_b3[32];
    __shared__ float s_sum[32];

    const int t = threadIdx.x;
    const int b = blockIdx.y;
    const long long bond0 = (long long)blockIdx.x * 128;

    // Stage weights transposed + converted to f16 (B-operand friendly layout).
    for (int i = t; i < 64 * 128; i += TB) { int n = i >> 7, k = i & 127; s_wt1[n * 128 + k] = (_Float16)w1[k * 64 + n]; }
    for (int i = t; i < 64 * 64;  i += TB) { int n = i >> 6, k = i & 63;  s_wt2[n * 64 + k]  = (_Float16)w2[k * 64 + n]; }
    for (int i = t; i < 32 * 64;  i += TB) { int n = i >> 6, k = i & 63;  s_wt3[n * 64 + k]  = (_Float16)w3[k * 32 + n]; }
    if (t < 64) { s_b1[t] = b1[t]; s_b2[t] = b2[t]; }
    if (t < 32) { s_b3[t] = b3[t]; s_sum[t] = 0.0f; }
    if (t < 128) {
        s_i1[t] = ba1[(long long)b * NB + bond0 + t];
        s_i2[t] = ba2[(long long)b * NB + bond0 + t];
    }
    __syncthreads();

    // Stage e_in = [atoms[a1] | atoms[a2] | bonds | state]  (128 features)
    for (int i = t; i < 128 * 128; i += TB) {
        int row = i >> 7, f = i & 127;
        float v;
        if (f < 32)       v = atoms[((long long)b * NA + s_i1[row]) * FDIM + f];
        else if (f < 64)  v = atoms[((long long)b * NA + s_i2[row]) * FDIM + (f - 32)];
        else if (f < 96)  v = bonds[((long long)b * NB + bond0 + row) * FDIM + (f - 64)];
        else              v = state[b * FDIM + (f - 96)];
        s_ein[i] = (_Float16)v;
    }
    __syncthreads();

    const int wave = t >> 5, lane = t & 31;
    const int grp = lane >> 4, nl = lane & 15;
    const int row0 = wave * 16;

    // ---- layer 1: [16x128] x [128x64] ----
    v8f acc1[4];
    for (int n = 0; n < 4; ++n) {
        float bv = s_b1[n * 16 + nl];
        for (int r = 0; r < 8; ++r) acc1[n][r] = bv;
    }
    for (int k = 0; k < 4; ++k) {
        v16h a = load_frag(&s_ein[(row0 + nl) * 128 + k * 32 + grp * 8]);
        for (int n = 0; n < 4; ++n) {
            v16h bm = load_frag(&s_wt1[(n * 16 + nl) * 128 + k * 32 + grp * 8]);
            acc1[n] = __builtin_amdgcn_wmma_f32_16x16x32_f16(false, a, false, bm, (short)0, acc1[n], false, false);
        }
    }
    for (int n = 0; n < 4; ++n)
        for (int r = 0; r < 8; ++r)
            s_act[(row0 + grp * 8 + r) * 64 + n * 16 + nl] = (_Float16)softplus_f(acc1[n][r]);

    // ---- layer 2: [16x64] x [64x64] ----
    v8f acc2[4];
    for (int n = 0; n < 4; ++n) {
        float bv = s_b2[n * 16 + nl];
        for (int r = 0; r < 8; ++r) acc2[n][r] = bv;
    }
    for (int k = 0; k < 2; ++k) {
        v16h a = load_frag(&s_act[(row0 + nl) * 64 + k * 32 + grp * 8]);
        for (int n = 0; n < 4; ++n) {
            v16h bm = load_frag(&s_wt2[(n * 16 + nl) * 64 + k * 32 + grp * 8]);
            acc2[n] = __builtin_amdgcn_wmma_f32_16x16x32_f16(false, a, false, bm, (short)0, acc2[n], false, false);
        }
    }
    for (int n = 0; n < 4; ++n)
        for (int r = 0; r < 8; ++r)
            s_act[(row0 + grp * 8 + r) * 64 + n * 16 + nl] = (_Float16)softplus_f(acc2[n][r]);

    // ---- layer 3: [16x64] x [64x32] ----
    v8f acc3[2];
    for (int n = 0; n < 2; ++n) {
        float bv = s_b3[n * 16 + nl];
        for (int r = 0; r < 8; ++r) acc3[n][r] = bv;
    }
    for (int k = 0; k < 2; ++k) {
        v16h a = load_frag(&s_act[(row0 + nl) * 64 + k * 32 + grp * 8]);
        for (int n = 0; n < 2; ++n) {
            v16h bm = load_frag(&s_wt3[(n * 16 + nl) * 64 + k * 32 + grp * 8]);
            acc3[n] = __builtin_amdgcn_wmma_f32_16x16x32_f16(false, a, false, bm, (short)0, acc3[n], false, false);
        }
    }

    // Write bonds_out, scatter-add into atom accumulator, accumulate mean sum.
    for (int n = 0; n < 2; ++n)
        for (int r = 0; r < 8; ++r) {
            int M = grp * 8 + r, col = n * 16 + nl;
            float v = softplus_f(acc3[n][r]);
            out_bonds[((long long)b * NB + bond0 + row0 + M) * FDIM + col] = v;
            int atom = s_i1[row0 + M];
            atomicAdd(&accA[((long long)b * NA + atom) * FDIM + col], v);
            atomicAdd(&s_sum[col], v);
        }
    if (lane < 16)
        atomicAdd(&cnt[(long long)b * NA + s_i1[row0 + lane]], 1.0f);

    __syncthreads();
    if (t < 32) atomicAdd(&bond_sum[b * 32 + t], s_sum[t]);
}

// ---------------------------------------------------------------------------
// Kernel 2: node MLP (96->64->64->32) over atoms, with mean-normalized
// bond aggregation; accumulates atom mean sums.
// ---------------------------------------------------------------------------
__global__ void __launch_bounds__(TB)
atom_kernel(const float* __restrict__ atoms, const float* __restrict__ state,
            const float* __restrict__ w1, const float* __restrict__ b1,
            const float* __restrict__ w2, const float* __restrict__ b2,
            const float* __restrict__ w3, const float* __restrict__ b3,
            const float* __restrict__ accA, const float* __restrict__ cnt,
            float* __restrict__ out_atoms, float* __restrict__ atom_sum)
{
    __shared__ __align__(16) _Float16 s_vin[128 * 96];
    __shared__ __align__(16) _Float16 s_wt1[64 * 96];
    __shared__ __align__(16) _Float16 s_wt2[64 * 64];
    __shared__ __align__(16) _Float16 s_wt3[32 * 64];
    __shared__ __align__(16) _Float16 s_act[128 * 64];
    __shared__ float s_b1[64], s_b2[64], s_b3[32];
    __shared__ float s_inv[128];
    __shared__ float s_sum[32];

    const int t = threadIdx.x;
    const int b = blockIdx.y;
    const int atom0 = blockIdx.x * 128;

    for (int i = t; i < 64 * 96; i += TB) { int n = i / 96, k = i % 96; s_wt1[n * 96 + k] = (_Float16)w1[k * 64 + n]; }
    for (int i = t; i < 64 * 64; i += TB) { int n = i >> 6, k = i & 63; s_wt2[n * 64 + k] = (_Float16)w2[k * 64 + n]; }
    for (int i = t; i < 32 * 64; i += TB) { int n = i >> 6, k = i & 63; s_wt3[n * 64 + k] = (_Float16)w3[k * 32 + n]; }
    if (t < 64) { s_b1[t] = b1[t]; s_b2[t] = b2[t]; }
    if (t < 32) { s_b3[t] = b3[t]; s_sum[t] = 0.0f; }
    if (t < 128) s_inv[t] = 1.0f / cnt[(long long)b * NA + atom0 + t];
    __syncthreads();

    // v_in = [bonds_to_atoms/counts | atoms | state]  (96 features)
    for (int i = t; i < 128 * 96; i += TB) {
        int row = i / 96, f = i % 96;
        float v;
        if (f < 32)       v = accA[((long long)b * NA + atom0 + row) * FDIM + f] * s_inv[row];
        else if (f < 64)  v = atoms[((long long)b * NA + atom0 + row) * FDIM + (f - 32)];
        else              v = state[b * FDIM + (f - 64)];
        s_vin[i] = (_Float16)v;
    }
    __syncthreads();

    const int wave = t >> 5, lane = t & 31;
    const int grp = lane >> 4, nl = lane & 15;
    const int row0 = wave * 16;

    v8f acc1[4];
    for (int n = 0; n < 4; ++n) {
        float bv = s_b1[n * 16 + nl];
        for (int r = 0; r < 8; ++r) acc1[n][r] = bv;
    }
    for (int k = 0; k < 3; ++k) {
        v16h a = load_frag(&s_vin[(row0 + nl) * 96 + k * 32 + grp * 8]);
        for (int n = 0; n < 4; ++n) {
            v16h bm = load_frag(&s_wt1[(n * 16 + nl) * 96 + k * 32 + grp * 8]);
            acc1[n] = __builtin_amdgcn_wmma_f32_16x16x32_f16(false, a, false, bm, (short)0, acc1[n], false, false);
        }
    }
    for (int n = 0; n < 4; ++n)
        for (int r = 0; r < 8; ++r)
            s_act[(row0 + grp * 8 + r) * 64 + n * 16 + nl] = (_Float16)softplus_f(acc1[n][r]);

    v8f acc2[4];
    for (int n = 0; n < 4; ++n) {
        float bv = s_b2[n * 16 + nl];
        for (int r = 0; r < 8; ++r) acc2[n][r] = bv;
    }
    for (int k = 0; k < 2; ++k) {
        v16h a = load_frag(&s_act[(row0 + nl) * 64 + k * 32 + grp * 8]);
        for (int n = 0; n < 4; ++n) {
            v16h bm = load_frag(&s_wt2[(n * 16 + nl) * 64 + k * 32 + grp * 8]);
            acc2[n] = __builtin_amdgcn_wmma_f32_16x16x32_f16(false, a, false, bm, (short)0, acc2[n], false, false);
        }
    }
    for (int n = 0; n < 4; ++n)
        for (int r = 0; r < 8; ++r)
            s_act[(row0 + grp * 8 + r) * 64 + n * 16 + nl] = (_Float16)softplus_f(acc2[n][r]);

    v8f acc3[2];
    for (int n = 0; n < 2; ++n) {
        float bv = s_b3[n * 16 + nl];
        for (int r = 0; r < 8; ++r) acc3[n][r] = bv;
    }
    for (int k = 0; k < 2; ++k) {
        v16h a = load_frag(&s_act[(row0 + nl) * 64 + k * 32 + grp * 8]);
        for (int n = 0; n < 2; ++n) {
            v16h bm = load_frag(&s_wt3[(n * 16 + nl) * 64 + k * 32 + grp * 8]);
            acc3[n] = __builtin_amdgcn_wmma_f32_16x16x32_f16(false, a, false, bm, (short)0, acc3[n], false, false);
        }
    }

    for (int n = 0; n < 2; ++n)
        for (int r = 0; r < 8; ++r) {
            int M = grp * 8 + r, col = n * 16 + nl;
            float v = softplus_f(acc3[n][r]);
            out_atoms[((long long)b * NA + atom0 + row0 + M) * FDIM + col] = v;
            atomicAdd(&s_sum[col], v);
        }

    __syncthreads();
    if (t < 32) atomicAdd(&atom_sum[b * 32 + t], s_sum[t]);
}

// ---------------------------------------------------------------------------
// Kernel 3: state MLP on 4 rows (tiny -> plain VALU).
// ---------------------------------------------------------------------------
__global__ void __launch_bounds__(TB)
state_kernel(const float* __restrict__ state,
             const float* __restrict__ bond_sum, const float* __restrict__ atom_sum,
             const float* __restrict__ w1, const float* __restrict__ b1,
             const float* __restrict__ w2, const float* __restrict__ b2,
             const float* __restrict__ w3, const float* __restrict__ b3,
             float* __restrict__ out_state)
{
    __shared__ float u_in[BATCH * 96];
    __shared__ float h1[BATCH * 64];
    __shared__ float h2[BATCH * 64];
    const int t = threadIdx.x;

    for (int i = t; i < BATCH * 96; i += TB) {
        int row = i / 96, f = i % 96;
        float v;
        if (f < 32)      v = bond_sum[row * 32 + f] * (1.0f / (float)NB);
        else if (f < 64) v = atom_sum[row * 32 + (f - 32)] * (1.0f / (float)NA);
        else             v = state[row * FDIM + (f - 64)];
        u_in[i] = v;
    }
    __syncthreads();

    {   // layer 1: 96 -> 64
        int row = t >> 6, j = t & 63;
        float s = b1[j];
        for (int k = 0; k < 96; ++k) s += u_in[row * 96 + k] * w1[k * 64 + j];
        h1[row * 64 + j] = softplus_f(s);
    }
    __syncthreads();
    {   // layer 2: 64 -> 64
        int row = t >> 6, j = t & 63;
        float s = b2[j];
        for (int k = 0; k < 64; ++k) s += h1[row * 64 + k] * w2[k * 64 + j];
        h2[row * 64 + j] = softplus_f(s);
    }
    __syncthreads();
    if (t < BATCH * 32) {   // layer 3: 64 -> 32
        int row = t >> 5, j = t & 31;
        float s = b3[j];
        for (int k = 0; k < 64; ++k) s += h2[row * 64 + k] * w3[k * 32 + j];
        out_state[row * 32 + j] = softplus_f(s);
    }
}

extern "C" void kernel_launch(void* const* d_in, const int* in_sizes, int n_in,
                              void* d_out, int out_size, void* d_ws, size_t ws_size,
                              hipStream_t stream) {
    const float* bonds = (const float*)d_in[0];
    const int*   ba1   = (const int*)d_in[1];
    const int*   ba2   = (const int*)d_in[2];
    const float* atoms = (const float*)d_in[3];
    const float* state = (const float*)d_in[4];
    const float* e_w1 = (const float*)d_in[5],  *e_b1 = (const float*)d_in[6];
    const float* e_w2 = (const float*)d_in[7],  *e_b2 = (const float*)d_in[8];
    const float* e_w3 = (const float*)d_in[9],  *e_b3 = (const float*)d_in[10];
    const float* v_w1 = (const float*)d_in[11], *v_b1 = (const float*)d_in[12];
    const float* v_w2 = (const float*)d_in[13], *v_b2 = (const float*)d_in[14];
    const float* v_w3 = (const float*)d_in[15], *v_b3 = (const float*)d_in[16];
    const float* u_w1 = (const float*)d_in[17], *u_b1 = (const float*)d_in[18];
    const float* u_w2 = (const float*)d_in[19], *u_b2 = (const float*)d_in[20];
    const float* u_w3 = (const float*)d_in[21], *u_b3 = (const float*)d_in[22];

    float* out = (float*)d_out;
    float* out_bonds = out;                                         // [B,NB,32]
    float* out_atoms = out + (long long)BATCH * NB * FDIM;          // [B,NA,32]
    float* out_state = out_atoms + (long long)BATCH * NA * FDIM;    // [B,1,32]

    float* accA     = (float*)d_ws;                                 // [B,NA,32]
    float* cnt      = accA + (long long)BATCH * NA * FDIM;          // [B,NA]
    float* bond_sum = cnt + (long long)BATCH * NA;                  // [B,32]
    float* atom_sum = bond_sum + BATCH * 32;                        // [B,32]
    unsigned long long nzero = (unsigned long long)BATCH * NA * FDIM
                             + (unsigned long long)BATCH * NA + 2ull * BATCH * 32;

    zero_kernel<<<(unsigned)((nzero + TB - 1) / TB), TB, 0, stream>>>((float*)d_ws, nzero);

    bond_kernel<<<dim3(NB / 128, BATCH), TB, 0, stream>>>(
        bonds, ba1, ba2, atoms, state,
        e_w1, e_b1, e_w2, e_b2, e_w3, e_b3,
        out_bonds, accA, cnt, bond_sum);

    atom_kernel<<<dim3(NA / 128, BATCH), TB, 0, stream>>>(
        atoms, state, v_w1, v_b1, v_w2, v_b2, v_w3, v_b3,
        accA, cnt, out_atoms, atom_sum);

    state_kernel<<<1, TB, 0, stream>>>(
        state, bond_sum, atom_sum,
        u_w1, u_b1, u_w2, u_b2, u_w3, u_b3, out_state);
}
